// MHA_90649579750090
// MI455X (gfx1250) — compile-verified, matrix operand order checked by reading, not run on hardware
//
#include <hip/hip_runtime.h>

// ---------------------------------------------------------------------------
// Types
// ---------------------------------------------------------------------------
typedef __bf16 bf16_t;
typedef __attribute__((ext_vector_type(16))) __bf16 v16bf;
typedef __attribute__((ext_vector_type(8)))  __bf16 v8bf;
typedef __attribute__((ext_vector_type(4)))  __bf16 v4bf;
typedef __attribute__((ext_vector_type(8)))  float  v8f;
typedef __attribute__((ext_vector_type(4)))  unsigned int v4u;
typedef __attribute__((ext_vector_type(8)))  int v8i;
typedef __attribute__((ext_vector_type(4)))  int v4i;

// Problem constants (match reference)
constexpr int   Bn       = 4;
constexpr int   Sn       = 4096;
constexpr int   Dn       = 1024;
constexpr float QK_SCALE = 0.03125f;   // 1/sqrt(1024)

// Padded LDS strides (bank-conflict-free fragment reads)
constexpr int KROWB   = 2064;          // 2048B row + 2x8B TDM pad  (=516 dw == 4 mod 64)
constexpr int KTILEB  = 32 * KROWB;    // 66048 B per K tile
constexpr int VROWB   = 80;            // 64B row + 16B TDM pad     (=20 dw)
constexpr int VTILEB  = 1024 * VROWB;  // 81920 B per V tile

// TDM group1 control words: data_size=2B (<<16), pad_enable(<<20),
// pad_interval(<<22, code: 7=256dw=1024B, 3=16dw=64B), pad_amount(<<25, code+1 dwords)
constexpr unsigned CTRL_K = 0x00010000u | (1u << 20) | (7u << 22) | (1u << 25); // +8B/1024B
constexpr unsigned CTRL_V = 0x00010000u | (1u << 20) | (3u << 22) | (3u << 25); // +16B/64B

#if defined(__has_builtin)
#  if __has_builtin(__builtin_amdgcn_tensor_load_to_lds) && \
      __has_builtin(__builtin_amdgcn_s_wait_tensorcnt)
#    define HAVE_TDM 1
#  endif
#endif
#ifndef HAVE_TDM
#  define HAVE_TDM 0
#endif

// ---------------------------------------------------------------------------
// WMMA helper: D = A(16x32 bf16) * B(32x16 bf16) + C(16x16 f32)
// ---------------------------------------------------------------------------
__device__ inline v8f wmma_bf16(v16bf a, v16bf b, v8f c) {
  return __builtin_amdgcn_wmma_f32_16x16x32_bf16(
      false, a, false, b, (short)0, c, false, false);
}

// Load a 16x32 bf16 tile in WMMA A/B-layout from a row-major matrix with a
// BYTE row stride (works for global memory and padded LDS tiles).
// lanes 0-15: row m, K={0..7,16..23}; lanes 16-31: row m, K={8..15,24..31}.
__device__ inline v16bf load_frag_b(const void* p, int stride_b) {
  const int lane = threadIdx.x & 31;
  const char* r = (const char*)p + (size_t)(lane & 15) * stride_b +
                  ((lane & 16) ? 16 : 0);
  v8bf lo = *(const v8bf*)(r);
  v8bf hi = *(const v8bf*)(r + 32);
  v16bf f;
#pragma unroll
  for (int i = 0; i < 8; ++i) { f[i] = lo[i]; f[i + 8] = hi[i]; }
  return f;
}

__device__ inline void prefetch_rows32(const bf16_t* p, int ld) {
  const int lane = threadIdx.x & 31;
  __builtin_prefetch(p + (size_t)lane * ld, 0, 0);
}

__device__ inline unsigned lds_addr_of(const void* p) {
  return (unsigned)(size_t)p;   // low 32 bits of flat LDS pointer = LDS offset
}

// ---------------------------------------------------------------------------
// TDM: async 2-D tile load global->LDS (Tensor Data Mover, TENSORcnt-tracked)
// ---------------------------------------------------------------------------
#if HAVE_TDM
__device__ inline void tdm_load_2d(unsigned lds_off, const void* gaddr,
                                   unsigned tensor_d0, unsigned tensor_d1,
                                   unsigned tile_d0, unsigned tile_d1,
                                   unsigned d0_stride, unsigned ctrl) {
  unsigned long long ga = (unsigned long long)gaddr;
  v4u g0;
  g0[0] = 1u;                                   // count=1, user load descriptor
  g0[1] = lds_off;                              // LDS byte address
  g0[2] = (unsigned)ga;                         // global_addr[31:0]
  g0[3] = (unsigned)((ga >> 32) & 0x1FFFFFFull) | (2u << 30);  // [56:32] | type=2
  v8i g1;
  g1[0] = (int)ctrl;                                            // mask/ds/pad
  g1[1] = (int)((tensor_d0 & 0xFFFFu) << 16);                   // tdim0[15:0]
  g1[2] = (int)((tensor_d0 >> 16) | ((tensor_d1 & 0xFFFFu) << 16));
  g1[3] = (int)((tensor_d1 >> 16) | ((tile_d0 & 0xFFFFu) << 16));
  g1[4] = (int)(tile_d1 & 0xFFFFu);                             // tile_dim1, dim2=0
  g1[5] = (int)d0_stride;                                       // dim0 stride lo32
  g1[6] = 0;
  g1[7] = 0;
  v4i gz = {0, 0, 0, 0};
#if defined(__clang_major__) && (__clang_major__ >= 23)
  v8i gz8 = {0, 0, 0, 0, 0, 0, 0, 0};
  __builtin_amdgcn_tensor_load_to_lds(g0, g1, gz, gz, gz8, 0);
#else
  __builtin_amdgcn_tensor_load_to_lds(g0, g1, gz, gz, 0);
#endif
}
#else
// Fallback: cooperative copy into the same padded LDS layouts.
__device__ inline void coop_copy_K(char* dst, const bf16_t* src, int nthr, int tid) {
  for (int c = tid; c < 32 * 128; c += nthr) {          // 16B chunks
    int r = c >> 7, wb = (c & 127) * 16;
    const char* s = (const char*)(src + (size_t)r * Dn) + wb;
    char* d = dst + (size_t)r * KROWB + wb + ((wb >= 1024) ? 8 : 0);
    ((unsigned long long*)d)[0] = ((const unsigned long long*)s)[0];
    ((unsigned long long*)d)[1] = ((const unsigned long long*)s)[1];
  }
}
__device__ inline void coop_copy_V(char* dst, const bf16_t* src, int nthr, int tid) {
  for (int c = tid; c < 1024 * 4; c += nthr) {          // 16B chunks
    int r = c >> 2, wb = (c & 3) * 16;
    const char* s = (const char*)(src + (size_t)r * Sn) + wb;
    char* d = dst + (size_t)r * VROWB + wb;
    ((unsigned long long*)d)[0] = ((const unsigned long long*)s)[0];
    ((unsigned long long*)d)[1] = ((const unsigned long long*)s)[1];
  }
}
#endif

// ---------------------------------------------------------------------------
// fp32 -> bf16 conversion (4 elements / thread)
// ---------------------------------------------------------------------------
__global__ void cvt_f32_bf16(const float* __restrict__ in,
                             bf16_t* __restrict__ out, int n) {
  int i = (blockIdx.x * blockDim.x + threadIdx.x) * 4;
  if (i < n) {
    float4 f = *(const float4*)(in + i);
    v4bf o;
    o[0] = (bf16_t)f.x; o[1] = (bf16_t)f.y;
    o[2] = (bf16_t)f.z; o[3] = (bf16_t)f.w;
    *(v4bf*)(out + i) = o;
  }
}

// ---------------------------------------------------------------------------
// C[M,N] = A[M,K] @ W[N,K]^T + bias (WMMA bf16, f32 accum)
// MODE 0: bf16 row-major; MODE 1: bf16 transposed [b][N][s]; MODE 2: f32.
// 8 waves; wave tile 64x64 (4x4 accums); block tile 256x128.
// ---------------------------------------------------------------------------
template <int MODE>
__device__ inline void store_tile(void* __restrict__ outp,
                                  const float* __restrict__ bias,
                                  const v8f& c, int row0, int col0, int ldN) {
  const int lane = threadIdx.x & 31;
  const int off  = (lane & 16) ? 8 : 0;
  const int col  = col0 + (lane & 15);
  const float bv = bias[col];
  if constexpr (MODE == 0) {
    bf16_t* o = (bf16_t*)outp;
#pragma unroll
    for (int j = 0; j < 8; ++j)
      o[(size_t)(row0 + off + j) * ldN + col] = (bf16_t)(c[j] + bv);
  } else if constexpr (MODE == 1) {
    bf16_t* o = (bf16_t*)outp;
    const int m  = row0 + off;
    const int bi = m / Sn;
    const int s  = m - bi * Sn;
    v8bf v;
#pragma unroll
    for (int j = 0; j < 8; ++j) v[j] = (bf16_t)(c[j] + bv);
    *(v8bf*)(o + ((size_t)bi * Dn + col) * Sn + s) = v;
  } else {
    float* o = (float*)outp;
#pragma unroll
    for (int j = 0; j < 8; ++j)
      o[(size_t)(row0 + off + j) * ldN + col] = c[j] + bv;
  }
}

template <int MODE>
__global__ __launch_bounds__(256) void gemm_wmma(
    const bf16_t* __restrict__ A, const bf16_t* __restrict__ W,
    const float* __restrict__ bias, void* __restrict__ Cout,
    int M, int N, int K) {
  const int w  = threadIdx.x >> 5;
  const int m0 = blockIdx.x * 256 + (w & 3) * 64;
  const int n0 = blockIdx.y * 128 + (w >> 2) * 64;

  v8f acc[4][4];
#pragma unroll
  for (int i = 0; i < 4; ++i)
#pragma unroll
    for (int j = 0; j < 4; ++j) acc[i][j] = {};

  for (int k = 0; k < K; k += 32) {
    if (k + 64 < K) {
      prefetch_rows32(A + (size_t)m0 * K + k + 64, K);
      prefetch_rows32(W + (size_t)n0 * K + k + 64, K);
    }
    v16bf a[4], b[4];
#pragma unroll
    for (int i = 0; i < 4; ++i)
      a[i] = load_frag_b(A + (size_t)(m0 + 16 * i) * K + k, K * 2);
#pragma unroll
    for (int j = 0; j < 4; ++j)
      b[j] = load_frag_b(W + (size_t)(n0 + 16 * j) * K + k, K * 2);
#pragma unroll
    for (int i = 0; i < 4; ++i)
#pragma unroll
      for (int j = 0; j < 4; ++j)
        acc[i][j] = wmma_bf16(a[i], b[j], acc[i][j]);
  }
#pragma unroll
  for (int i = 0; i < 4; ++i)
#pragma unroll
    for (int j = 0; j < 4; ++j)
      store_tile<MODE>(Cout, bias, acc[i][j], m0 + 16 * i, n0 + 16 * j, N);
}

// ---------------------------------------------------------------------------
// Pass 1: softmax stats (running max m, sum l).  4 waves x 16 query rows.
// K tiles (32x1024 bf16) staged in LDS by TDM, double-buffered.
// ---------------------------------------------------------------------------
__global__ __launch_bounds__(128) void attn_stats(
    const bf16_t* __restrict__ Q, const bf16_t* __restrict__ Kc,
    float* __restrict__ mbuf, float* __restrict__ lbuf) {
  __shared__ alignas(16) char Ksh[2][KTILEB];

  const int w    = threadIdx.x >> 5;
  const int lane = threadIdx.x & 31;
  const int b    = blockIdx.y;
  const int q0   = blockIdx.x * 64 + w * 16;

  const bf16_t* Qp = Q  + ((size_t)b * Sn + q0) * Dn;
  const bf16_t* Kp = Kc + (size_t)b * Sn * Dn;

  float mr[8], ls[8];
#pragma unroll
  for (int j = 0; j < 8; ++j) { mr[j] = -1e30f; ls[j] = 0.0f; }

#if HAVE_TDM
  if (w == 0)
    tdm_load_2d(lds_addr_of(Ksh[0]), Kp, Dn, 32, Dn, 32, Dn, CTRL_K);
#endif

  for (int kt = 0; kt < Sn; kt += 32) {
    const int p = (kt >> 5) & 1;
#if HAVE_TDM
    if (w == 0) {
      if (kt + 32 < Sn) {
        tdm_load_2d(lds_addr_of(Ksh[p ^ 1]), Kp + (size_t)(kt + 32) * Dn,
                    Dn, 32, Dn, 32, Dn, CTRL_K);
        __builtin_amdgcn_s_wait_tensorcnt(1);
      } else {
        __builtin_amdgcn_s_wait_tensorcnt(0);
      }
    }
#else
    coop_copy_K(Ksh[p], Kp + (size_t)kt * Dn, 128, threadIdx.x);
#endif
    __syncthreads();

    const char* Kt = Ksh[p];
    v8f c0 = {}, c1 = {};
    for (int kc = 0; kc < Dn; kc += 32) {
      const int kq = kc * 2 + ((kc & 512) ? 8 : 0);    // padded byte offset
      v16bf a  = load_frag_b(Qp + kc, Dn * 2);
      v16bf b0 = load_frag_b(Kt + kq, KROWB);
      v16bf b1 = load_frag_b(Kt + 16 * KROWB + kq, KROWB);
      c0 = wmma_bf16(a, b0, c0);
      c1 = wmma_bf16(a, b1, c1);
    }
#pragma unroll
    for (int j = 0; j < 8; ++j) {
      float s0 = c0[j] * QK_SCALE;
      float s1 = c1[j] * QK_SCALE;
      float mx = fmaxf(s0, s1);
#pragma unroll
      for (int d = 1; d < 16; d <<= 1) mx = fmaxf(mx, __shfl_xor(mx, d, 32));
      float nm = fmaxf(mr[j], mx);
      float e  = __expf(s0 - nm) + __expf(s1 - nm);
#pragma unroll
      for (int d = 1; d < 16; d <<= 1) e += __shfl_xor(e, d, 32);
      ls[j] = ls[j] * __expf(mr[j] - nm) + e;
      mr[j] = nm;
    }
    __syncthreads();    // all waves done with Ksh[p^1]'s successor before refill
  }
  const int off = (lane & 16) ? 8 : 0;
  if ((lane & 15) == 0) {
#pragma unroll
    for (int j = 0; j < 8; ++j) {
      mbuf[(size_t)b * Sn + q0 + off + j] = mr[j];
      lbuf[(size_t)b * Sn + q0 + off + j] = ls[j];
    }
  }
}

// ---------------------------------------------------------------------------
// Pass 2: Z = softmax(Q K^T * scale) @ V.  8 waves = 4 query-groups x 2
// D-halves; K (32x1024) and V (1024x32, from Vt[b][D][S]) tiles staged by
// TDM, double-buffered; P transposed C->A layout through LDS.
// ---------------------------------------------------------------------------
__global__ __launch_bounds__(256) void attn_apply(
    const bf16_t* __restrict__ Q, const bf16_t* __restrict__ Kc,
    const bf16_t* __restrict__ Vt, const float* __restrict__ mbuf,
    const float* __restrict__ lbuf, bf16_t* __restrict__ Z) {
  __shared__ alignas(16) char Ksh[2][KTILEB];    // 132096 B
  __shared__ alignas(16) char Vsh[2][VTILEB];    // 163840 B
  __shared__ alignas(16) bf16_t Pst[4][16 * 32]; //   4096 B

  const int w    = threadIdx.x >> 5;
  const int lane = threadIdx.x & 31;
  const int g    = w >> 1;
  const int h    = w & 1;
  const int b    = blockIdx.y;
  const int q0   = blockIdx.x * 64 + g * 16;
  const int off  = (lane & 16) ? 8 : 0;

  const bf16_t* Qp = Q  + ((size_t)b * Sn + q0) * Dn;
  const bf16_t* Kp = Kc + (size_t)b * Sn * Dn;
  const bf16_t* Vp = Vt + (size_t)b * Dn * Sn;

  float mrow[8], ilrow[8];
#pragma unroll
  for (int j = 0; j < 8; ++j) {
    const size_t r = (size_t)b * Sn + q0 + off + j;
    mrow[j]  = mbuf[r];
    ilrow[j] = 1.0f / lbuf[r];
  }

  v8f z[32];
#pragma unroll
  for (int i = 0; i < 32; ++i) z[i] = {};

  bf16_t* ps = &Pst[g][0];
  const int n0 = h * 512;

#if HAVE_TDM
  if (w == 0) {
    tdm_load_2d(lds_addr_of(Ksh[0]), Kp, Dn, 32, Dn, 32, Dn, CTRL_K);
    tdm_load_2d(lds_addr_of(Vsh[0]), Vp, 32, Dn, 32, Dn, Sn, CTRL_V);
  }
#endif

  for (int kt = 0; kt < Sn; kt += 32) {
    const int p = (kt >> 5) & 1;
#if HAVE_TDM
    if (w == 0) {
      if (kt + 32 < Sn) {
        tdm_load_2d(lds_addr_of(Ksh[p ^ 1]), Kp + (size_t)(kt + 32) * Dn,
                    Dn, 32, Dn, 32, Dn, CTRL_K);
        tdm_load_2d(lds_addr_of(Vsh[p ^ 1]), Vp + kt + 32,
                    32, Dn, 32, Dn, Sn, CTRL_V);
        __builtin_amdgcn_s_wait_tensorcnt(2);
      } else {
        __builtin_amdgcn_s_wait_tensorcnt(0);
      }
    }
#else
    coop_copy_K(Ksh[p], Kp + (size_t)kt * Dn, 256, threadIdx.x);
    coop_copy_V(Vsh[p], Vp + kt, 256, threadIdx.x);
#endif
    __syncthreads();

    const char* Kt = Ksh[p];
    const char* Vtile = Vsh[p];

    // Scores for this wave's 16-key half (LDS K, conflict-free padded rows).
    v8f c = {};
    for (int kc = 0; kc < Dn; kc += 32) {
      const int kq = kc * 2 + ((kc & 512) ? 8 : 0);
      v16bf a  = load_frag_b(Qp + kc, Dn * 2);
      v16bf bb = load_frag_b(Kt + (size_t)(h * 16) * KROWB + kq, KROWB);
      c = wmma_bf16(a, bb, c);
    }
    // P = exp(s*scale - m) / l, staged row-major 16x32 in LDS
#pragma unroll
    for (int j = 0; j < 8; ++j) {
      float pv = __expf(c[j] * QK_SCALE - mrow[j]) * ilrow[j];
      ps[(off + j) * 32 + h * 16 + (lane & 15)] = (bf16_t)pv;
    }
    __syncthreads();
    v16bf pa = load_frag_b(ps, 64);   // A-layout read of full 16x32 P tile
#pragma unroll
    for (int nt = 0; nt < 32; ++nt) {
      v16bf vf = load_frag_b(Vtile + (size_t)(n0 + nt * 16) * VROWB, VROWB);
      z[nt] = wmma_bf16(pa, vf, z[nt]);
    }
    __syncthreads();                  // Pst + K/V buffers reused next iter
  }

  // Store Z (bf16 row-major [B*S][D]); 1/l already folded into P.
#pragma unroll
  for (int nt = 0; nt < 32; ++nt) {
    const int col = n0 + nt * 16 + (lane & 15);
#pragma unroll
    for (int j = 0; j < 8; ++j)
      Z[((size_t)b * Sn + q0 + off + j) * Dn + col] = (bf16_t)z[nt][j];
  }
}

// ---------------------------------------------------------------------------
// Launcher
// ---------------------------------------------------------------------------
extern "C" void kernel_launch(void* const* d_in, const int* in_sizes, int n_in,
                              void* d_out, int out_size, void* d_ws,
                              size_t ws_size, hipStream_t stream) {
  (void)in_sizes; (void)n_in; (void)out_size; (void)ws_size;
  const float* x  = (const float*)d_in[0];
  const float* Wq = (const float*)d_in[1];
  const float* bq = (const float*)d_in[2];
  const float* Wk = (const float*)d_in[3];
  const float* bk = (const float*)d_in[4];
  const float* Wv = (const float*)d_in[5];
  const float* bv = (const float*)d_in[6];
  const float* Wo = (const float*)d_in[7];
  const float* bo = (const float*)d_in[8];
  float* out = (float*)d_out;

  const size_t BSD = (size_t)Bn * Sn * Dn;
  const size_t DD  = (size_t)Dn * Dn;

  char* ws = (char*)d_ws;
  size_t off = 0;
  auto take = [&](size_t bytes) -> char* {
    char* p = ws + off;
    off += (bytes + 255) & ~(size_t)255;
    return p;
  };
  bf16_t* xb  = (bf16_t*)take(BSD * 2);
  bf16_t* Wqb = (bf16_t*)take(DD * 2);
  bf16_t* Wkb = (bf16_t*)take(DD * 2);
  bf16_t* Wvb = (bf16_t*)take(DD * 2);
  bf16_t* Wob = (bf16_t*)take(DD * 2);
  bf16_t* Qb  = (bf16_t*)take(BSD * 2);
  bf16_t* Kb  = (bf16_t*)take(BSD * 2);
  bf16_t* Vtb = (bf16_t*)take(BSD * 2);      // [B][D][S]
  bf16_t* Zb  = (bf16_t*)take(BSD * 2);
  float*  mb  = (float*)take((size_t)Bn * Sn * 4);
  float*  lb  = (float*)take((size_t)Bn * Sn * 4);

  cvt_f32_bf16<<<(int)((BSD / 4 + 255) / 256), 256, 0, stream>>>(x, xb, (int)BSD);
  cvt_f32_bf16<<<(int)((DD / 4 + 255) / 256), 256, 0, stream>>>(Wq, Wqb, (int)DD);
  cvt_f32_bf16<<<(int)((DD / 4 + 255) / 256), 256, 0, stream>>>(Wk, Wkb, (int)DD);
  cvt_f32_bf16<<<(int)((DD / 4 + 255) / 256), 256, 0, stream>>>(Wv, Wvb, (int)DD);
  cvt_f32_bf16<<<(int)((DD / 4 + 255) / 256), 256, 0, stream>>>(Wo, Wob, (int)DD);

  const int M = Bn * Sn;
  dim3 gg(M / 256, Dn / 128);
  gemm_wmma<0><<<gg, 256, 0, stream>>>(xb, Wqb, bq, Qb, M, Dn, Dn);
  gemm_wmma<0><<<gg, 256, 0, stream>>>(xb, Wkb, bk, Kb, M, Dn, Dn);
  gemm_wmma<1><<<gg, 256, 0, stream>>>(xb, Wvb, bv, Vtb, M, Dn, Dn);

  dim3 ga(Sn / 64, Bn);
  attn_stats<<<ga, 128, 0, stream>>>(Qb, Kb, mb, lb);
  attn_apply<<<ga, 256, 0, stream>>>(Qb, Kb, Vtb, mb, lb, Zb);

  gemm_wmma<2><<<gg, 256, 0, stream>>>(Zb, Wob, bo, out, M, Dn, Dn);
}
